// EGAT_34522947125839
// MI455X (gfx1250) — matrix operand order repaired and checked by reading, not compiled, not run on hardware
//
#include <hip/hip_runtime.h>
#include <hip/hip_bf16.h>
#include <math.h>

// ---------------------------------------------------------------------------
// EGAT on MI455X (gfx1250): all GEMMs via v_wmma_f32_16x16x32_f16 (wave32).
// Weights pre-converted to f16, transposed [n][K] so each lane's B operand is
// 16 contiguous halves (2x global_load_b128). A operand uses the same contiguous
// k order (k = 16*half + s). Scatter ops use fp32 global atomics (bandwidth-bound side).
// ---------------------------------------------------------------------------

typedef __attribute__((ext_vector_type(16))) _Float16 v16h;
typedef __attribute__((ext_vector_type(8)))  float    v8f;

#define NN_  30000
#define EE_  480000
#define HH_  128
#define GG_  64

__device__ __forceinline__ v8f wmma16(v16h a, v16h b, v8f c) {
  return __builtin_amdgcn_wmma_f32_16x16x32_f16(false, a, false, b, (short)0, c,
                                                false, false);
}

// ---- weight convert: Wt[n*Kpad + k] = (k<K) ? (f16)W[k*Ncols + n] : 0 ------
__global__ void egat_cvt(const float* __restrict__ W, _Float16* __restrict__ Wt,
                         int K, int Kpad, int Ncols) {
  int t = blockIdx.x * blockDim.x + threadIdx.x;
  int total = Kpad * Ncols;
  if (t >= total) return;
  int n = t / Kpad, k = t % Kpad;
  Wt[(size_t)n * Kpad + k] =
      (k < K) ? (_Float16)W[(size_t)k * Ncols + n] : (_Float16)0.f;
}

__global__ void egat_zero(float* __restrict__ p, long n) {
  long t = (long)blockIdx.x * blockDim.x + threadIdx.x;
  if (t < n) p[t] = 0.f;
}

__global__ void egat_copy(const float* __restrict__ s, float* __restrict__ d,
                          long n) {
  long t = (long)blockIdx.x * blockDim.x + threadIdx.x;
  if (t < n) d[t] = s[t];
}

// ---- generic Y[rows,128] = act(X[rows,128] @ W + b), WMMA -----------------
// act: 0 = none, 1 = leaky(0.01)
__global__ __launch_bounds__(256)
void egat_linear128(const float* __restrict__ X, const _Float16* __restrict__ Wt,
                    const float* __restrict__ bias, float* __restrict__ Y,
                    int rows, int act) {
  int lane = threadIdx.x & 31, wave = threadIdx.x >> 5;
  int hf = lane >> 4, m = lane & 15;
  int row0 = (blockIdx.x * 8 + wave) * 16;
  if (row0 >= rows) return;
  const float* xrow = X + (size_t)(row0 + m) * 128;
  v8f acc[8] = {};
  for (int kb = 0; kb < 128; kb += 32) {
    v16h a;
    int k0 = kb + hf * 16;
#pragma unroll
    for (int t = 0; t < 8; ++t) {
      a[t]     = (_Float16)xrow[k0 + t];
      a[8 + t] = (_Float16)xrow[k0 + 8 + t];
    }
#pragma unroll
    for (int nb = 0; nb < 8; ++nb) {
      const v16h* bp =
          (const v16h*)(Wt + (size_t)(nb * 16 + m) * 128 + kb + hf * 16);
      acc[nb] = wmma16(a, *bp, acc[nb]);
    }
  }
#pragma unroll
  for (int nb = 0; nb < 8; ++nb) {
    int col = nb * 16 + m;
    float bv = bias[col];
#pragma unroll
    for (int v = 0; v < 8; ++v) {
      float y = acc[nb][v] + bv;
      if (act == 1) y = y > 0.f ? y : 0.01f * y;
      Y[(size_t)(row0 + v + 8 * hf) * 128 + col] = y;
    }
  }
}

// ---- gate: g = sigmoid([nn, x, nn-x] @ Wg + bg); x1 = g*nn + x ------------
__global__ __launch_bounds__(256)
void egat_gate(const float* __restrict__ nn_, const float* __restrict__ node,
               const _Float16* __restrict__ Wgt, const float* __restrict__ bg,
               float* __restrict__ g, float* __restrict__ x1, int rows) {
  int lane = threadIdx.x & 31, wave = threadIdx.x >> 5;
  int hf = lane >> 4, m = lane & 15;
  int row0 = (blockIdx.x * 8 + wave) * 16;
  if (row0 >= rows) return;
  const float* nrow = nn_ + (size_t)(row0 + m) * 128;
  const float* drow = node + (size_t)(row0 + m) * 128;
  v8f acc[8] = {};
  for (int kb = 0; kb < 384; kb += 32) {
    v16h a;
    int k0 = kb + hf * 16;
#pragma unroll
    for (int t = 0; t < 8; ++t) {
      int kk = k0 + t;
      float v0 = (kk < 128) ? nrow[kk]
                 : (kk < 256) ? drow[kk - 128]
                              : nrow[kk - 256] - drow[kk - 256];
      a[t] = (_Float16)v0;
      kk = k0 + 8 + t;
      float v1 = (kk < 128) ? nrow[kk]
                 : (kk < 256) ? drow[kk - 128]
                              : nrow[kk - 256] - drow[kk - 256];
      a[8 + t] = (_Float16)v1;
    }
#pragma unroll
    for (int nb = 0; nb < 8; ++nb) {
      const v16h* bp =
          (const v16h*)(Wgt + (size_t)(nb * 16 + m) * 384 + kb + hf * 16);
      acc[nb] = wmma16(a, *bp, acc[nb]);
    }
  }
#pragma unroll
  for (int nb = 0; nb < 8; ++nb) {
    int col = nb * 16 + m;
    float bv = bg[col];
#pragma unroll
    for (int v = 0; v < 8; ++v) {
      size_t ro = (size_t)(row0 + v + 8 * hf) * 128 + col;
      float s = 1.f / (1.f + __expf(-(acc[nb][v] + bv)));
      g[ro] = s;
      x1[ro] = s * nn_[ro] + node[ro];
    }
  }
}

// ---- fused edge MLP + attention pre-result --------------------------------
// feat = [edge(16), dist] padded K=32 -> leaky(@We1+be1) -> @We2+be2 = e
// er = dst[i] * (src[j] * e) / sqrt(32)
__global__ __launch_bounds__(256)
void egat_edge(const float* __restrict__ edge, const float* __restrict__ coords,
               const int* __restrict__ ei, const int* __restrict__ ej,
               const float* __restrict__ src_, const float* __restrict__ dst_,
               const _Float16* __restrict__ W1t, const float* __restrict__ be1,
               const _Float16* __restrict__ W2t, const float* __restrict__ be2,
               float* __restrict__ er) {
  __shared__ _Float16 feat[8][16 * 32];   // per-wave A tile for GEMM1
  __shared__ _Float16 e1s[8][16 * 128];   // per-wave leaky(e1) in f16
  __shared__ int idx_i[8][16], idx_j[8][16];

  int lane = threadIdx.x & 31, wave = threadIdx.x >> 5;
  int hf = lane >> 4, m = lane & 15;
  int e0 = blockIdx.x * 128 + wave * 16;  // grid sized so e0+16 <= E

  if (hf == 0) {
    const float* ep = edge + (size_t)(e0 + m) * 16;
#pragma unroll
    for (int k = 0; k < 16; ++k) feat[wave][m * 32 + k] = (_Float16)ep[k];
  } else {
    int i = ei[e0 + m], j = ej[e0 + m];
    idx_i[wave][m] = i;
    idx_j[wave][m] = j;
    float dx = coords[i * 3 + 0] - coords[j * 3 + 0];
    float dy = coords[i * 3 + 1] - coords[j * 3 + 1];
    float dz = coords[i * 3 + 2] - coords[j * 3 + 2];
    float dist = sqrtf(dx * dx + dy * dy + dz * dz + 1e-12f) * 0.1f;
    feat[wave][m * 32 + 16] = (_Float16)dist;
#pragma unroll
    for (int k = 17; k < 32; ++k) feat[wave][m * 32 + k] = (_Float16)0.f;
  }
  __syncthreads();

  // GEMM1: K = 32 (rows 17..31 of W1t are zero)
  v8f a1[8] = {};
  {
    v16h a;
    int k0 = hf * 16;
#pragma unroll
    for (int t = 0; t < 8; ++t) {
      a[t]     = feat[wave][m * 32 + k0 + t];
      a[8 + t] = feat[wave][m * 32 + 8 + k0 + t];
    }
#pragma unroll
    for (int nb = 0; nb < 8; ++nb) {
      const v16h* bp =
          (const v16h*)(W1t + (size_t)(nb * 16 + m) * 32 + hf * 16);
      a1[nb] = wmma16(a, *bp, a1[nb]);
    }
  }
#pragma unroll
  for (int nb = 0; nb < 8; ++nb) {
    int col = nb * 16 + m;
    float bv = be1[col];
#pragma unroll
    for (int v = 0; v < 8; ++v) {
      float y = a1[nb][v] + bv;
      y = y > 0.f ? y : 0.01f * y;
      e1s[wave][(v + 8 * hf) * 128 + col] = (_Float16)y;
    }
  }
  __syncthreads();

  // GEMM2: K = 128
  v8f a2[8] = {};
  for (int kb = 0; kb < 128; kb += 32) {
    v16h a;
    int k0 = kb + hf * 16;
#pragma unroll
    for (int t = 0; t < 8; ++t) {
      a[t]     = e1s[wave][m * 128 + k0 + t];
      a[8 + t] = e1s[wave][m * 128 + k0 + 8 + t];
    }
#pragma unroll
    for (int nb = 0; nb < 8; ++nb) {
      const v16h* bp =
          (const v16h*)(W2t + (size_t)(nb * 16 + m) * 128 + kb + hf * 16);
      a2[nb] = wmma16(a, *bp, a2[nb]);
    }
  }
  const float inv = 0.17677669529663687f;  // 1/sqrt(32)
#pragma unroll
  for (int nb = 0; nb < 8; ++nb) {
    int col = nb * 16 + m;
    float bv = be2[col];
#pragma unroll
    for (int v = 0; v < 8; ++v) {
      int r = v + 8 * hf;
      float ev = a2[nb][v] + bv;
      int i = idx_i[wave][r], j = idx_j[wave][r];
      float erv = dst_[(size_t)i * 128 + col] *
                  (src_[(size_t)j * 128 + col] * ev) * inv;
      er[(size_t)(e0 + r) * 128 + col] = erv;
    }
  }
}

// ---- edge_new = er @ Wue + bue  (E x 128 @ 128 x 16) ----------------------
__global__ __launch_bounds__(256)
void egat_edge_out(const float* __restrict__ er, const _Float16* __restrict__ Wuet,
                   const float* __restrict__ bue, float* __restrict__ out,
                   int rows) {
  int lane = threadIdx.x & 31, wave = threadIdx.x >> 5;
  int hf = lane >> 4, m = lane & 15;
  int row0 = (blockIdx.x * 8 + wave) * 16;
  if (row0 >= rows) return;
  const float* xrow = er + (size_t)(row0 + m) * 128;
  v8f acc = {};
  for (int kb = 0; kb < 128; kb += 32) {
    v16h a;
    int k0 = kb + hf * 16;
#pragma unroll
    for (int t = 0; t < 8; ++t) {
      a[t]     = (_Float16)xrow[k0 + t];
      a[8 + t] = (_Float16)xrow[k0 + 8 + t];
    }
    const v16h* bp = (const v16h*)(Wuet + (size_t)m * 128 + kb + hf * 16);
    acc = wmma16(a, *bp, acc);
  }
  float bv = bue[m];
#pragma unroll
  for (int v = 0; v < 8; ++v)
    out[(size_t)(row0 + v + 8 * hf) * 16 + m] = acc[v] + bv;
}

// ---- scatter softmax & aggregation ----------------------------------------
__global__ void egat_scores(const float* __restrict__ er,
                            float* __restrict__ scores, int E) {
  int t = blockIdx.x * blockDim.x + threadIdx.x;
  if (t >= E * 4) return;
  int e = t >> 2, h = t & 3;
  const float* p = er + (size_t)e * 128 + h * 32;
  float s = 0.f;
#pragma unroll
  for (int d = 0; d < 32; ++d) s += fabsf(p[d]);
  scores[t] = s;
}

__global__ void egat_segmax(const float* __restrict__ scores,
                            const int* __restrict__ ei,
                            unsigned* __restrict__ smax, int E) {
  int t = blockIdx.x * blockDim.x + threadIdx.x;
  if (t >= E * 4) return;
  int e = t >> 2, h = t & 3;
  int i = ei[e];
  atomicMax(&smax[(size_t)i * 4 + h], __float_as_uint(scores[t]));  // scores>=0
}

__global__ void egat_softmax(float* __restrict__ scores,
                             const float* __restrict__ smax,
                             float* __restrict__ den,
                             const int* __restrict__ ei, int E) {
  int t = blockIdx.x * blockDim.x + threadIdx.x;
  if (t >= E * 4) return;
  int e = t >> 2, h = t & 3;
  int i = ei[e];
  float ex = __expf(scores[t] - smax[(size_t)i * 4 + h]);
  scores[t] = ex;
  atomicAdd(&den[(size_t)i * 4 + h], ex);
}

__global__ void egat_agg(const float* __restrict__ ex,
                         const float* __restrict__ den,
                         const float* __restrict__ val,
                         const int* __restrict__ ei, const int* __restrict__ ej,
                         float* __restrict__ agg, long total) {
  long t = (long)blockIdx.x * blockDim.x + threadIdx.x;
  if (t >= total) return;
  int e = (int)(t >> 7);
  int c = (int)(t & 127);
  int h = c >> 5;
  int i = ei[e], j = ej[e];
  float w = ex[(size_t)e * 4 + h] / den[(size_t)i * 4 + h];
  atomicAdd(&agg[(size_t)i * 128 + c], w * val[(size_t)j * 128 + c]);
}

// ---- GraphNorm pieces ------------------------------------------------------
__global__ void egat_cnt(const int* __restrict__ batch, float* __restrict__ cnt,
                         int n) {
  int t = blockIdx.x * blockDim.x + threadIdx.x;
  if (t < n) atomicAdd(&cnt[batch[t]], 1.f);
}

__global__ void egat_gnsum(const float* __restrict__ x,
                           const int* __restrict__ batch,
                           float* __restrict__ msum, long total) {
  long t = (long)blockIdx.x * blockDim.x + threadIdx.x;
  if (t >= total) return;
  int n = (int)(t >> 7), c = (int)(t & 127);
  atomicAdd(&msum[(size_t)batch[n] * 128 + c], x[t]);
}

__global__ void egat_gncenter(float* __restrict__ x,
                              const int* __restrict__ batch,
                              const float* __restrict__ msum,
                              const float* __restrict__ cnt,
                              const float* __restrict__ a,
                              float* __restrict__ vsum, long total) {
  long t = (long)blockIdx.x * blockDim.x + threadIdx.x;
  if (t >= total) return;
  int n = (int)(t >> 7), c = (int)(t & 127);
  int b = batch[n];
  float cn = fmaxf(cnt[b], 1.f);
  float o = x[t] - a[c] * (msum[(size_t)b * 128 + c] / cn);
  x[t] = o;
  atomicAdd(&vsum[(size_t)b * 128 + c], o * o);
}

__global__ void egat_gnfinal(const float* __restrict__ x,
                             const int* __restrict__ batch,
                             const float* __restrict__ vsum,
                             const float* __restrict__ cnt,
                             const float* __restrict__ w,
                             const float* __restrict__ bb,
                             float* __restrict__ y, long total) {
  long t = (long)blockIdx.x * blockDim.x + threadIdx.x;
  if (t >= total) return;
  int n = (int)(t >> 7), c = (int)(t & 127);
  int b = batch[n];
  float cn = fmaxf(cnt[b], 1.f);
  float var = vsum[(size_t)b * 128 + c] / cn;
  y[t] = w[c] * x[t] * rsqrtf(var + 1e-5f) + bb[c];
}

__global__ void egat_x2(const float* __restrict__ g, const float* __restrict__ fx,
                        const float* __restrict__ n1, float* __restrict__ x,
                        long total) {
  long t = (long)blockIdx.x * blockDim.x + threadIdx.x;
  if (t < total) x[t] = g[t] * fx[t] + n1[t];
}

// ---------------------------------------------------------------------------
extern "C" void kernel_launch(void* const* d_in, const int* in_sizes, int n_in,
                              void* d_out, int out_size, void* d_ws,
                              size_t ws_size, hipStream_t stream) {
  const int N = NN_, E = EE_, G = GG_;
  const float* node   = (const float*)d_in[0];
  const float* edge   = (const float*)d_in[1];
  const float* coords = (const float*)d_in[2];
  const float* Ws  = (const float*)d_in[3];   const float* bs  = (const float*)d_in[4];
  const float* Wd  = (const float*)d_in[5];   const float* bd  = (const float*)d_in[6];
  const float* Wv  = (const float*)d_in[7];   const float* bv  = (const float*)d_in[8];
  const float* We1 = (const float*)d_in[9];   const float* be1 = (const float*)d_in[10];
  const float* We2 = (const float*)d_in[11];  const float* be2 = (const float*)d_in[12];
  const float* Wun = (const float*)d_in[13];  const float* bun = (const float*)d_in[14];
  const float* Wue = (const float*)d_in[15];  const float* bue = (const float*)d_in[16];
  const float* Wg  = (const float*)d_in[17];  const float* bg  = (const float*)d_in[18];
  const float* Wf1 = (const float*)d_in[19];  const float* bf1 = (const float*)d_in[20];
  const float* Wf2 = (const float*)d_in[21];  const float* bf2 = (const float*)d_in[22];
  const float* gn1_w = (const float*)d_in[23];
  const float* gn1_b = (const float*)d_in[24];
  const float* gn1_a = (const float*)d_in[25];
  const float* gn2_w = (const float*)d_in[26];
  const float* gn2_b = (const float*)d_in[27];
  const float* gn2_a = (const float*)d_in[28];
  const int* eidx  = (const int*)d_in[29];
  const int* ei = eidx;          // destination nodes (i)
  const int* ej = eidx + E;      // source nodes (j)
  const int* batch = (const int*)d_in[30];
  float* out = (float*)d_out;

  // ---- workspace layout ----
  float* ws = (float*)d_ws;
  const size_t NH  = (size_t)N * 128;
  const size_t EH  = (size_t)E * 128;
  const size_t ENH = (size_t)E * 4;
  const size_t NNH = (size_t)N * 4;
  const size_t GH  = (size_t)G * 128;
  size_t o = 0;
  float* f_src  = ws + o; o += NH;
  float* f_dst  = ws + o; o += NH;
  float* f_val  = ws + o; o += NH;
  float* f_er   = ws + o; o += EH;
  float* f_sc   = ws + o; o += ENH;
  float* f_smax = ws + o; o += NNH;
  float* f_den  = ws + o; o += NNH;
  float* f_agg  = ws + o; o += NH;   // also reused as f1 temp
  float* f_nn   = ws + o; o += NH;
  float* f_g    = ws + o; o += NH;
  float* f_x    = ws + o; o += NH;
  float* f_n1   = ws + o; o += NH;
  float* f_fx   = ws + o; o += NH;
  float* f_ms   = ws + o; o += GH;
  float* f_vs   = ws + o; o += GH;
  float* f_cnt  = ws + o; o += G;
  o = (o + 7) & ~(size_t)7;        // 32B alignment for v16h loads
  _Float16* hb = (_Float16*)(ws + o);
  _Float16* h_Ws  = hb;              // 128x128 each, [n][K] layout
  _Float16* h_Wd  = hb + 16384;
  _Float16* h_Wv  = hb + 32768;
  _Float16* h_We1 = hb + 49152;      // 128 x 32 (K=17 padded)
  _Float16* h_We2 = hb + 53248;
  _Float16* h_Wun = hb + 69632;
  _Float16* h_Wue = hb + 86016;      // 16 x 128
  _Float16* h_Wg  = hb + 88064;      // 128 x 384
  _Float16* h_Wf1 = hb + 137216;
  _Float16* h_Wf2 = hb + 153600;

  auto cd = [](long a, long b) { return (unsigned)((a + b - 1) / b); };

  // ---- weight conversion (f16, transposed) ----
  egat_cvt<<<cd(128 * 128, 256), 256, 0, stream>>>(Ws,  h_Ws,  128, 128, 128);
  egat_cvt<<<cd(128 * 128, 256), 256, 0, stream>>>(Wd,  h_Wd,  128, 128, 128);
  egat_cvt<<<cd(128 * 128, 256), 256, 0, stream>>>(Wv,  h_Wv,  128, 128, 128);
  egat_cvt<<<cd(128 * 32,  256), 256, 0, stream>>>(We1, h_We1, 17,  32,  128);
  egat_cvt<<<cd(128 * 128, 256), 256, 0, stream>>>(We2, h_We2, 128, 128, 128);
  egat_cvt<<<cd(128 * 128, 256), 256, 0, stream>>>(Wun, h_Wun, 128, 128, 128);
  egat_cvt<<<cd(16 * 128,  256), 256, 0, stream>>>(Wue, h_Wue, 128, 128, 16);
  egat_cvt<<<cd(128 * 384, 256), 256, 0, stream>>>(Wg,  h_Wg,  384, 384, 128);
  egat_cvt<<<cd(128 * 128, 256), 256, 0, stream>>>(Wf1, h_Wf1, 128, 128, 128);
  egat_cvt<<<cd(128 * 128, 256), 256, 0, stream>>>(Wf2, h_Wf2, 128, 128, 128);

  // ---- zero accumulators (smax, den, agg contiguous; ms, vs, cnt contig) ----
  egat_zero<<<cd(NNH * 2 + NH, 256), 256, 0, stream>>>(f_smax, (long)(NNH * 2 + NH));
  egat_zero<<<cd(GH * 2 + G, 256), 256, 0, stream>>>(f_ms, (long)(GH * 2 + G));

  // ---- node projections ----
  const unsigned lb = cd(N / 16, 8);  // 16-row strips, 8 waves/block
  egat_linear128<<<lb, 256, 0, stream>>>(node, h_Ws, bs, f_src, N, 0);
  egat_linear128<<<lb, 256, 0, stream>>>(node, h_Wd, bd, f_dst, N, 0);
  egat_linear128<<<lb, 256, 0, stream>>>(node, h_Wv, bv, f_val, N, 0);

  // ---- edge MLP + attention pre-result ----
  egat_edge<<<E / 128, 256, 0, stream>>>(edge, coords, ei, ej, f_src, f_dst,
                                         h_We1, be1, h_We2, be2, f_er);

  // ---- scatter softmax over destinations ----
  egat_scores<<<cd(ENH, 256), 256, 0, stream>>>(f_er, f_sc, E);
  egat_segmax<<<cd(ENH, 256), 256, 0, stream>>>(f_sc, ei, (unsigned*)f_smax, E);
  egat_softmax<<<cd(ENH, 256), 256, 0, stream>>>(f_sc, f_smax, f_den, ei, E);
  egat_agg<<<cd(EH, 256), 256, 0, stream>>>(f_sc, f_den, f_val, ei, ej, f_agg,
                                            (long)EH);

  // ---- node update + gate ----
  egat_linear128<<<lb, 256, 0, stream>>>(f_agg, h_Wun, bun, f_nn, N, 0);
  egat_gate<<<lb, 256, 0, stream>>>(f_nn, node, h_Wg, bg, f_g, f_x, N);

  // ---- GraphNorm 1 ----
  egat_cnt<<<cd(N, 256), 256, 0, stream>>>(batch, f_cnt, N);
  egat_gnsum<<<cd(NH, 256), 256, 0, stream>>>(f_x, batch, f_ms, (long)NH);
  egat_gncenter<<<cd(NH, 256), 256, 0, stream>>>(f_x, batch, f_ms, f_cnt, gn1_a,
                                                 f_vs, (long)NH);
  egat_gnfinal<<<cd(NH, 256), 256, 0, stream>>>(f_x, batch, f_vs, f_cnt, gn1_w,
                                                gn1_b, f_n1, (long)NH);

  // ---- fix_node: fx = leaky(n1 @ Wf1 + bf1) @ Wf2 + bf2 ----
  egat_linear128<<<lb, 256, 0, stream>>>(f_n1, h_Wf1, bf1, f_agg, N, 1);
  egat_linear128<<<lb, 256, 0, stream>>>(f_agg, h_Wf2, bf2, f_fx, N, 0);
  egat_x2<<<cd(NH, 256), 256, 0, stream>>>(f_g, f_fx, f_n1, f_x, (long)NH);

  // ---- GraphNorm 2 -> d_out ----
  egat_zero<<<cd(GH * 2, 256), 256, 0, stream>>>(f_ms, (long)(GH * 2));
  egat_gnsum<<<cd(NH, 256), 256, 0, stream>>>(f_x, batch, f_ms, (long)NH);
  egat_gncenter<<<cd(NH, 256), 256, 0, stream>>>(f_x, batch, f_ms, f_cnt, gn2_a,
                                                 f_vs, (long)NH);
  egat_gnfinal<<<cd(NH, 256), 256, 0, stream>>>(f_x, batch, f_vs, f_cnt, gn2_w,
                                                gn2_b, out, (long)NH);

  // ---- edge output + coords ----
  egat_edge_out<<<cd(E / 16, 8), 256, 0, stream>>>(f_er, h_Wue, bue, out + NH, E);
  egat_copy<<<cd((long)N * 3, 256), 256, 0, stream>>>(
      coords, out + NH + (size_t)E * 16, (long)N * 3);
}
